// MoEFFN_32177894981778
// MI455X (gfx1250) — matrix operand order repaired and checked by reading, not compiled, not run on hardware
//
#include <hip/hip_runtime.h>

// ---- problem constants (match reference) ----
#define DDIM  2048   // model dim
#define NTOK  2048   // B*T tokens
#define NEXP  64     // experts
#define TOPK  4
#define FDIM  512    // expert hidden
#define NSH   2      // shared experts
#define SFDIM 2048   // shared hidden
#define CAP   256    // per-expert capacity
#define PSLOT (NTOK * TOPK)

typedef __attribute__((ext_vector_type(16))) __bf16 v16bf;
typedef __attribute__((ext_vector_type(8)))  float  v8f;

// ------------------------------------------------------------------
// WMMA fragment loaders.  CDNA5 16-bit A (16x32, MxK) lane layout:
//   lane l (group g=l>>4, m=l&15):  a[0..7]  = K = g*8 + 0..7
//                                   a[8..15] = K = 16 + g*8 + 0..7
// B (32x16, KxN) lane layout:
//   lane l (g=l>>4, n=l&15):        b[0..7] = K = g*8 + 0..7, b[8..15] = K = 16 + g*8 + 0..7 (same as A)
// Both are contiguous per-lane runs in row-major [row, K] storage.
// ------------------------------------------------------------------
static __device__ __forceinline__ v16bf load_a_f32(const float* __restrict__ rowk, int lane) {
  const float* p = rowk + ((lane >> 4) * 8);
  v16bf a;
#pragma unroll
  for (int i = 0; i < 8; ++i) { a[i] = (__bf16)p[i]; a[8 + i] = (__bf16)p[16 + i]; }
  return a;
}

static __device__ __forceinline__ v16bf load_a_bf16(const __bf16* __restrict__ rowk, int lane) {
  const __bf16* p = rowk + ((lane >> 4) * 8);
  v16bf a;
#pragma unroll
  for (int i = 0; i < 8; ++i) { a[i] = p[i]; a[8 + i] = p[16 + i]; }
  return a;
}

static __device__ __forceinline__ v16bf load_b_f32(const float* __restrict__ rowk, int lane) {
  const float* p = rowk + ((lane >> 4) * 8);
  v16bf b;
#pragma unroll
  for (int i = 0; i < 8; ++i) { b[i] = (__bf16)p[i]; b[8 + i] = (__bf16)p[16 + i]; }
  return b;
}

static __device__ __forceinline__ v8f wmma_bf16(v16bf a, v16bf b, v8f c) {
  return __builtin_amdgcn_wmma_f32_16x16x32_bf16(false, a, false, b, (short)0, c, false, false);
}

static __device__ __forceinline__ v8f vzero8() {
  v8f z = {0.f, 0.f, 0.f, 0.f, 0.f, 0.f, 0.f, 0.f};
  return z;
}

// ------------------------------------------------------------------
// 1) Router: logits = x @ router_w^T ; top-4 on (logits+bias), weights =
//    softmax over the 4 selected raw logits (== renormalized softmax scores).
// One block per token, 64 threads = 1 expert each.
// ------------------------------------------------------------------
__global__ void router_kernel(const float* __restrict__ x,
                              const float* __restrict__ rw,
                              const float* __restrict__ rb,
                              int* __restrict__ tki, float* __restrict__ tkw) {
  const int t = blockIdx.x;
  const int e = threadIdx.x;
  const float4* xp = (const float4*)(x + (size_t)t * DDIM);
  const float4* wp = (const float4*)(rw + (size_t)e * DDIM);
  float acc = 0.f;
  for (int k = 0; k < DDIM / 4; ++k) {
    float4 a = xp[k], b = wp[k];
    acc += a.x * b.x + a.y * b.y + a.z * b.z + a.w * b.w;
  }
  __shared__ float lg[NEXP];
  lg[e] = acc;
  __syncthreads();
  if (e == 0) {
    int   idx[TOPK];
    float val[TOPK];
    bool  used[NEXP];
    for (int i = 0; i < NEXP; ++i) used[i] = false;
    for (int j = 0; j < TOPK; ++j) {
      float best = -3.4e38f; int bi = 0;
      for (int i = 0; i < NEXP; ++i) {           // ascending scan + strict '>' == JAX tie-break
        if (!used[i]) {
          float v = lg[i] + rb[i];
          if (v > best) { best = v; bi = i; }
        }
      }
      used[bi] = true; idx[j] = bi; val[j] = lg[bi];   // weight from raw logit
    }
    float m = val[0];
    for (int j = 1; j < TOPK; ++j) m = fmaxf(m, val[j]);
    float w[TOPK]; float s = 0.f;
    for (int j = 0; j < TOPK; ++j) { w[j] = __expf(val[j] - m); s += w[j]; }
    float inv = 1.f / s;
    for (int j = 0; j < TOPK; ++j) {
      tki[t * TOPK + j] = idx[j];
      tkw[t * TOPK + j] = w[j] * inv;
    }
  }
}

// ------------------------------------------------------------------
// 2) Dispatch: stable sort-by-expert via in-order wave scan + ballot prefix.
// One wave (32 threads) per expert.
// ------------------------------------------------------------------
__global__ void dispatch_kernel(const int* __restrict__ tki,
                                const float* __restrict__ tkw,
                                int* __restrict__ tok_tab,
                                float* __restrict__ w_tab) {
  const int e = blockIdx.x;
  const int lane = threadIdx.x;
  int* tt = tok_tab + (size_t)e * CAP;
  float* wt = w_tab + (size_t)e * CAP;
  int base = 0;
  const unsigned lmask = (1u << lane) - 1u;
  for (int p0 = 0; p0 < PSLOT; p0 += 32) {
    int p = p0 + lane;
    bool match = (tki[p] == e);
    unsigned mask = __builtin_amdgcn_ballot_w32(match);
    int pos = base + __popc(mask & lmask);
    if (match && pos < CAP) {
      tt[pos] = p >> 2;           // token id
      wt[pos] = tkw[p];
    }
    base += __popc(mask);
  }
  int fill = base < CAP ? base : CAP;
  for (int s = fill + lane; s < CAP; s += 32) { tt[s] = 0; wt[s] = 0.f; }
}

// ------------------------------------------------------------------
// 3) Fused gate/up GEMM + SiLU:  H = silu(X Wg^T) * (X Wu^T), bf16 out.
// One wave per 32(M) x 32(N) tile.  Optional token gather for A rows.
// Batched B-fragment loads + k-unroll 2 for in-wave pipelining.
// ------------------------------------------------------------------
__global__ void __launch_bounds__(32, 1)
gemm_gateup(const float* __restrict__ X,
            const int* __restrict__ tok,     // nullable
            const float* __restrict__ Wg,
            const float* __restrict__ Wu,
            __bf16* __restrict__ H,
            int Kdim,
            long wBatchStride, long hBatchStride,
            int ldh, int tokBatchStride) {
  const int lane  = threadIdx.x;
  const int batch = blockIdx.x;
  const int m0    = blockIdx.y * 32;
  const int n0    = blockIdx.z * 32;

  const float* wg = Wg + (size_t)batch * wBatchStride;
  const float* wu = Wu + (size_t)batch * wBatchStride;
  __bf16* h = H + (size_t)batch * hBatchStride;

  const int r0 = m0 + (lane & 15);
  const int r1 = r0 + 16;
  size_t row0, row1;
  if (tok) {
    const int* tb = tok + (size_t)batch * tokBatchStride;
    row0 = (size_t)tb[r0]; row1 = (size_t)tb[r1];
  } else {
    row0 = (size_t)r0; row1 = (size_t)r1;
  }
  const float* arow0 = X + row0 * DDIM;
  const float* arow1 = X + row1 * DDIM;

  const int ncol = n0 + (lane & 15);
  const float* grow0 = wg + (size_t)ncol * Kdim;
  const float* grow1 = wg + (size_t)(ncol + 16) * Kdim;
  const float* urow0 = wu + (size_t)ncol * Kdim;
  const float* urow1 = wu + (size_t)(ncol + 16) * Kdim;

  v8f accg[2][2], accu[2][2];
#pragma unroll
  for (int s = 0; s < 2; ++s)
#pragma unroll
    for (int j = 0; j < 2; ++j) { accg[s][j] = vzero8(); accu[s][j] = vzero8(); }

#pragma unroll 2
  for (int k0 = 0; k0 < Kdim; k0 += 32) {
    // issue the whole k-step's loads up front (one big clause)
    v16bf a0  = load_a_f32(arow0 + k0, lane);
    v16bf a1  = load_a_f32(arow1 + k0, lane);
    v16bf bg0 = load_b_f32(grow0 + k0, lane);
    v16bf bg1 = load_b_f32(grow1 + k0, lane);
    v16bf bu0 = load_b_f32(urow0 + k0, lane);
    v16bf bu1 = load_b_f32(urow1 + k0, lane);
    __builtin_prefetch(grow0 + k0 + 64, 0, 1);
    __builtin_prefetch(urow0 + k0 + 64, 0, 1);
    accg[0][0] = wmma_bf16(a0, bg0, accg[0][0]);
    accg[1][0] = wmma_bf16(a1, bg0, accg[1][0]);
    accg[0][1] = wmma_bf16(a0, bg1, accg[0][1]);
    accg[1][1] = wmma_bf16(a1, bg1, accg[1][1]);
    accu[0][0] = wmma_bf16(a0, bu0, accu[0][0]);
    accu[1][0] = wmma_bf16(a1, bu0, accu[1][0]);
    accu[0][1] = wmma_bf16(a0, bu1, accu[0][1]);
    accu[1][1] = wmma_bf16(a1, bu1, accu[1][1]);
  }

  // C/D layout: lane l, elem r -> M = m0 + 16*s + 8*(l>>4) + r, N = n0 + 16*j + (l&15)
#pragma unroll
  for (int s = 0; s < 2; ++s)
#pragma unroll
    for (int j = 0; j < 2; ++j)
#pragma unroll
      for (int r = 0; r < 8; ++r) {
        int m = m0 + s * 16 + ((lane >> 4) * 8) + r;
        int n = ncol + j * 16;
        float g = accg[s][j][r];
        float u = accu[s][j][r];
        float sig = 1.f / (1.f + __expf(-g));
        h[(size_t)m * ldh + n] = (__bf16)(g * sig * u);
      }
}

// ------------------------------------------------------------------
// 4) Expert down GEMM + gate-weighted scatter-add: out[tok] += (H Wd^T) * w
// One wave per 32(M) x 64(N over DDIM) tile.
// ------------------------------------------------------------------
__global__ void __launch_bounds__(32, 1)
expert_down(const __bf16* __restrict__ H,
            const float* __restrict__ Wd,
            const int* __restrict__ tok,
            const float* __restrict__ wtab,
            float* __restrict__ out) {
  const int lane = threadIdx.x;
  const int e  = blockIdx.x;
  const int m0 = blockIdx.y * 32;
  const int n0 = blockIdx.z * 64;          // over DDIM

  const __bf16* h  = H + (size_t)e * CAP * FDIM;
  const float*  wd = Wd + (size_t)e * DDIM * FDIM;
  const int r0 = m0 + (lane & 15);
  const int r1 = r0 + 16;
  const int ncol = n0 + (lane & 15);
  const __bf16* h0 = h + (size_t)r0 * FDIM;
  const __bf16* h1 = h + (size_t)r1 * FDIM;

  v8f acc[2][4];
#pragma unroll
  for (int s = 0; s < 2; ++s)
#pragma unroll
    for (int j = 0; j < 4; ++j) acc[s][j] = vzero8();

#pragma unroll 2
  for (int k0 = 0; k0 < FDIM; k0 += 32) {
    v16bf a0 = load_a_bf16(h0 + k0, lane);
    v16bf a1 = load_a_bf16(h1 + k0, lane);
    v16bf b[4];
#pragma unroll
    for (int j = 0; j < 4; ++j)
      b[j] = load_b_f32(wd + (size_t)(ncol + j * 16) * FDIM + k0, lane);
#pragma unroll
    for (int j = 0; j < 4; ++j) {
      acc[0][j] = wmma_bf16(a0, b[j], acc[0][j]);
      acc[1][j] = wmma_bf16(a1, b[j], acc[1][j]);
    }
  }

  const int*   tb = tok  + (size_t)e * CAP;
  const float* wt = wtab + (size_t)e * CAP;
#pragma unroll
  for (int s = 0; s < 2; ++s)
#pragma unroll
    for (int j = 0; j < 4; ++j)
#pragma unroll
      for (int r = 0; r < 8; ++r) {
        int m = m0 + s * 16 + ((lane >> 4) * 8) + r;
        int d = ncol + j * 16;
        int t = tb[m];
        float w = wt[m];
        atomicAdd(out + (size_t)t * DDIM + d, acc[s][j][r] * w);
      }
}

// ------------------------------------------------------------------
// 5) Shared-experts down GEMM (sum over S in-register) + plain add.
// One wave per 32(M) x 64(N over DDIM) tile.
// ------------------------------------------------------------------
__global__ void __launch_bounds__(32, 1)
shared_down(const __bf16* __restrict__ Hsh,  // [NTOK, NSH*SFDIM]
            const float* __restrict__ Wd,    // [NSH, DDIM, SFDIM]
            float* __restrict__ out) {
  const int lane = threadIdx.x;
  const int m0 = blockIdx.x * 32;
  const int n0 = blockIdx.y * 64;
  const int r0 = m0 + (lane & 15);
  const int r1 = r0 + 16;
  const int ncol = n0 + (lane & 15);

  v8f acc[2][4];
#pragma unroll
  for (int s = 0; s < 2; ++s)
#pragma unroll
    for (int j = 0; j < 4; ++j) acc[s][j] = vzero8();

  for (int sx = 0; sx < NSH; ++sx) {
    const __bf16* h0 = Hsh + (size_t)r0 * (NSH * SFDIM) + (size_t)sx * SFDIM;
    const __bf16* h1 = Hsh + (size_t)r1 * (NSH * SFDIM) + (size_t)sx * SFDIM;
    const float*  wd = Wd + (size_t)sx * DDIM * SFDIM;
#pragma unroll 2
    for (int k0 = 0; k0 < SFDIM; k0 += 32) {
      v16bf a0 = load_a_bf16(h0 + k0, lane);
      v16bf a1 = load_a_bf16(h1 + k0, lane);
      v16bf b[4];
#pragma unroll
      for (int j = 0; j < 4; ++j) {
        const float* br = wd + (size_t)(ncol + j * 16) * SFDIM + k0;
        __builtin_prefetch(br + 64, 0, 1);
        b[j] = load_b_f32(br, lane);
      }
#pragma unroll
      for (int j = 0; j < 4; ++j) {
        acc[0][j] = wmma_bf16(a0, b[j], acc[0][j]);
        acc[1][j] = wmma_bf16(a1, b[j], acc[1][j]);
      }
    }
  }

#pragma unroll
  for (int s = 0; s < 2; ++s)
#pragma unroll
    for (int j = 0; j < 4; ++j)
#pragma unroll
      for (int r = 0; r < 8; ++r) {
        int m = m0 + s * 16 + ((lane >> 4) * 8) + r;
        int d = ncol + j * 16;
        size_t o = (size_t)m * DDIM + d;
        out[o] = out[o] + acc[s][j][r];   // exclusive owner; runs after scatter kernel
      }
}

// ------------------------------------------------------------------
extern "C" void kernel_launch(void* const* d_in, const int* in_sizes, int n_in,
                              void* d_out, int out_size, void* d_ws, size_t ws_size,
                              hipStream_t stream) {
  const float* x      = (const float*)d_in[0];
  const float* rw     = (const float*)d_in[1];
  const float* rb     = (const float*)d_in[2];
  const float* gate_w = (const float*)d_in[3];
  const float* up_w   = (const float*)d_in[4];
  const float* down_w = (const float*)d_in[5];
  const float* shg    = (const float*)d_in[6];
  const float* shu    = (const float*)d_in[7];
  const float* shd    = (const float*)d_in[8];
  float* out = (float*)d_out;

  char* p = (char*)d_ws;
  auto carve = [&](size_t bytes) -> char* {
    char* r = p;
    p += (bytes + 255) & ~(size_t)255;
    return r;
  };
  int*    topk_idx = (int*)   carve((size_t)NTOK * TOPK * sizeof(int));
  float*  topk_w   = (float*) carve((size_t)NTOK * TOPK * sizeof(float));
  int*    tok_tab  = (int*)   carve((size_t)NEXP * CAP * sizeof(int));
  float*  w_tab    = (float*) carve((size_t)NEXP * CAP * sizeof(float));
  __bf16* Hexp     = (__bf16*)carve((size_t)NEXP * CAP * FDIM * sizeof(__bf16));
  __bf16* Hsh      = (__bf16*)carve((size_t)NTOK * NSH * SFDIM * sizeof(__bf16));
  (void)ws_size; (void)in_sizes; (void)n_in;

  hipMemsetAsync(d_out, 0, (size_t)out_size * sizeof(float), stream);

  router_kernel<<<NTOK, NEXP, 0, stream>>>(x, rw, rb, topk_idx, topk_w);
  dispatch_kernel<<<NEXP, 32, 0, stream>>>(topk_idx, topk_w, tok_tab, w_tab);

  // routed experts: H = silu(Xg G^T) * (Xg U^T)
  gemm_gateup<<<dim3(NEXP, CAP / 32, FDIM / 32), 32, 0, stream>>>(
      x, tok_tab, gate_w, up_w, Hexp, DDIM,
      (long)FDIM * DDIM, (long)CAP * FDIM, FDIM, CAP);

  // shared experts: Hsh[n, s*SF+f]
  gemm_gateup<<<dim3(NSH, NTOK / 32, SFDIM / 32), 32, 0, stream>>>(
      x, nullptr, shg, shu, Hsh, DDIM,
      (long)SFDIM * DDIM, (long)SFDIM, NSH * SFDIM, 0);

  expert_down<<<dim3(NEXP, CAP / 32, DDIM / 64), 32, 0, stream>>>(
      Hexp, down_w, tok_tab, w_tab, out);

  shared_down<<<dim3(NTOK / 32, DDIM / 64), 32, 0, stream>>>(Hsh, shd, out);
}
